// GCN_88167088652543
// MI455X (gfx1250) — matrix operand order repaired and checked by reading, not compiled, not run on hardware
//
#include <hip/hip_runtime.h>
#include <hip/hip_bf16.h>

// ---- CDNA5 wave32 WMMA types ----
typedef __attribute__((ext_vector_type(16))) __bf16 v16bf;
typedef __attribute__((ext_vector_type(8)))  __bf16 v8bf;
typedef __attribute__((ext_vector_type(8)))  float  v8f;

#define TPB 256

// Zero a float buffer (d_ws is poisoned; AGG must be re-zeroed every call).
__global__ void k_zero(float* __restrict__ p, long n) {
    long i = (long)blockIdx.x * blockDim.x + threadIdx.x;
    if (i < n) p[i] = 0.0f;
}

// Degree counting: deg_out[src]++, deg_in[dst]++ (float atomics; exact for counts).
__global__ void k_degrees(const int* __restrict__ src, const int* __restrict__ dst,
                          float* __restrict__ dout, float* __restrict__ din, int ne) {
    int e = blockIdx.x * blockDim.x + threadIdx.x;
    if (e >= ne) return;
    atomicAdd(&dout[src[e]], 1.0f);
    atomicAdd(&din[dst[e]], 1.0f);
}

// norm = clamp(deg,1)^-0.5 in place.
__global__ void k_norms(float* __restrict__ dout, float* __restrict__ din, int n) {
    int i = blockIdx.x * blockDim.x + threadIdx.x;
    if (i >= n) return;
    float a = dout[i]; dout[i] = rsqrtf(a > 0.0f ? a : 1.0f);
    float b = din[i];  din[i]  = rsqrtf(b > 0.0f ? b : 1.0f);
}

// A_bf16[row,k] = bf16(H[row,k] * norm_out[row]); K fixed at 128.
__global__ void k_prep(const float* __restrict__ H, const float* __restrict__ norm_out,
                       __bf16* __restrict__ Abf, int nrows) {
    long t = (long)blockIdx.x * blockDim.x + threadIdx.x;
    if (t >= (long)nrows * 128) return;
    int row = (int)(t >> 7);
    Abf[t] = (__bf16)(H[t] * norm_out[row]);
}

// T[nrows x NOUT] = A[nrows x 128] (bf16) @ W[128 x NOUT] (f32 -> bf16 in LDS),
// f32 accumulation via v_wmma_f32_16x16x32_bf16.
// Block = 256 threads = 8 waves; each wave owns a 16-row tile -> 128 rows/block.
// NOUT is a template parameter so all strides/offsets are immediates and the
// store path is branch-free except the (compile-time known) ragged last tile.
template <int NOUT>
__global__ void __launch_bounds__(TPB)
k_gemm_wmma(const __bf16* __restrict__ A, const float* __restrict__ W,
            float* __restrict__ T, int nrows) {
    constexpr int NPAD   = (NOUT + 15) & ~15;
    constexpr int NTILES = NPAD / 16;
    // W transposed into LDS as bf16: Wt[n*128 + k], padded cols are zero.
    __shared__ __bf16 Wt[NPAD * 128];   // <= 32 KB of the 320 KB WGP LDS

    const int tid  = threadIdx.x;
    const int lane = tid & 31;
    const int wave = tid >> 5;
    const int half = lane >> 4;   // ISA 7.12.2: lanes 0-15 vs 16-31
    const int l16  = lane & 15;

    for (int idx = tid; idx < NPAD * 128; idx += TPB) {
        int n = idx >> 7;
        int k = idx & 127;
        Wt[idx] = (n < NOUT) ? (__bf16)W[k * NOUT + n] : (__bf16)0.0f;
    }
    __syncthreads();

    const int row0 = (blockIdx.x * 8 + wave) * 16;
    if (row0 >= nrows) return;
    int arow = row0 + l16;
    if (arow >= nrows) arow = nrows - 1;   // clamped load; stores are guarded
    const __bf16* Arow = A + (long)arow * 128;

    // A fragments (16x32 bf16): lane holds K in [k0+half*16, +16)
    v16bf afrag[4];
#pragma unroll
    for (int kf = 0; kf < 4; ++kf) {
        const int k0 = kf * 32 + half * 16;
        v8bf lo = *(const v8bf*)(Arow + k0);        // global_load_b128
        v8bf hi = *(const v8bf*)(Arow + k0 + 8);
#pragma unroll
        for (int i = 0; i < 8; ++i) { afrag[kf][i] = lo[i]; afrag[kf][8 + i] = hi[i]; }
    }

    // Rows are full whenever row0 < nrows and nrows % 16 == 0 (true for N=50000);
    // keep a wave-uniform scalar guard for generality.
    const bool rows_full = (row0 + 16 <= nrows);

#pragma unroll
    for (int nt = 0; nt < NTILES; ++nt) {
        const int col = nt * 16 + l16;
        v8f acc = {};
#pragma unroll
        for (int kf = 0; kf < 4; ++kf) {
            // B fragment (32x16 bf16): lane = column, K = kf*32 + half*16 + i
            const __bf16* bp = Wt + col * 128 + kf * 32 + half * 16;
            v8bf blo = *(const v8bf*)bp;            // ds_load_b128 (imm offsets)
            v8bf bhi = *(const v8bf*)(bp + 8);
            v16bf bfrag{};
#pragma unroll
            for (int i = 0; i < 8; ++i) { bfrag[i] = blo[i]; bfrag[8 + i] = bhi[i]; }
            // (neg_a, A, neg_b, B, c_mod, C, reuse_a, reuse_b)
            acc = __builtin_amdgcn_wmma_f32_16x16x32_bf16(
                false, afrag[kf], false, bfrag, (short)0, acc, false, false);
        }
        // C/D layout: VGPR r -> row = row0 + r + half*8, col = nt*16 + l16.
        const bool tile_full = ((nt + 1) * 16 <= NOUT);   // constant after unroll
        float* out = T + (long)(row0 + half * 8) * NOUT + col;
        if (rows_full) {
            if (tile_full) {
#pragma unroll
                for (int r = 0; r < 8; ++r) out[(long)r * NOUT] = acc[r];
            } else if (col < NOUT) {    // one exec mask around all 8 stores
#pragma unroll
                for (int r = 0; r < 8; ++r) out[(long)r * NOUT] = acc[r];
            }
        } else {
#pragma unroll
            for (int r = 0; r < 8; ++r) {
                const int row = row0 + r + half * 8;
                if (row < nrows && col < NOUT) T[(long)row * NOUT + col] = acc[r];
            }
        }
    }
}

// segment_sum: AGG[dst] += T[src], float4 chunk per thread, fp32 atomics
// (T and AGG are L2-resident at these sizes). Consecutive threads cover
// consecutive 16B chunks of one edge's row -> coalesced gather.
template <int NOUT>
__global__ void k_aggregate(const float* __restrict__ T, const int* __restrict__ src,
                            const int* __restrict__ dst, float* __restrict__ AGG,
                            int ne) {
    constexpr int CHUNKS = NOUT / 4;    // 32 or 10 (compile-time)
    long t = (long)blockIdx.x * blockDim.x + threadIdx.x;
    if (t >= (long)ne * CHUNKS) return;
    int e = (int)(t / CHUNKS);
    int c = (int)(t % CHUNKS) * 4;
    int s = src[e], d = dst[e];
    const float4 v = *(const float4*)(T + (long)s * NOUT + c);
    float* o = AGG + (long)d * NOUT + c;
    atomicAdd(o + 0, v.x);
    atomicAdd(o + 1, v.y);
    atomicAdd(o + 2, v.z);
    atomicAdd(o + 3, v.w);
}

// H = (AGG * norm_in[row]) + b[col], optional ReLU (compile-time).
template <int NOUT, bool RELU>
__global__ void k_finalize(const float* __restrict__ AGG, const float* __restrict__ norm_in,
                           const float* __restrict__ b, float* __restrict__ H,
                           int nrows) {
    long t = (long)blockIdx.x * blockDim.x + threadIdx.x;
    if (t >= (long)nrows * NOUT) return;
    int i = (int)(t / NOUT);
    int c = (int)(t % NOUT);
    float v = fmaf(AGG[t], norm_in[i], b[c]);
    H[t] = RELU ? fmaxf(v, 0.0f) : v;
}

extern "C" void kernel_launch(void* const* d_in, const int* in_sizes, int n_in,
                              void* d_out, int out_size, void* d_ws, size_t ws_size,
                              hipStream_t stream) {
    (void)n_in; (void)out_size; (void)ws_size;
    const float* feat = (const float*)d_in[0];
    const int*   src  = (const int*)d_in[1];
    const int*   dst  = (const int*)d_in[2];
    const float* W0   = (const float*)d_in[3];
    const float* b0   = (const float*)d_in[4];
    const float* W1   = (const float*)d_in[5];
    const float* b1   = (const float*)d_in[6];
    const float* W2   = (const float*)d_in[7];
    const float* b2   = (const float*)d_in[8];

    const int IN = 128, HID = 128, CLS = 40;
    const int N = in_sizes[0] / IN;
    const int E = in_sizes[1];

    // Workspace layout (256B aligned slices)
    char* ws = (char*)d_ws;
    size_t off = 0;
    auto alloc = [&](size_t bytes) -> void* {
        void* p = ws + off;
        off += (bytes + 255) & ~(size_t)255;
        return p;
    };
    float*  norm_out = (float*) alloc((size_t)N * 4);
    float*  norm_in  = (float*) alloc((size_t)N * 4);
    __bf16* Abf      = (__bf16*)alloc((size_t)N * 128 * 2);
    float*  T        = (float*) alloc((size_t)N * 128 * 4);
    float*  AGG      = (float*) alloc((size_t)N * 128 * 4);
    float*  H        = (float*) alloc((size_t)N * 128 * 4);

    auto blocks = [](long n) { return (int)((n + TPB - 1) / TPB); };

    // Degrees -> norms (recomputed every call; ws is not trusted across calls)
    k_zero   <<<blocks(N), TPB, 0, stream>>>(norm_out, (long)N);
    k_zero   <<<blocks(N), TPB, 0, stream>>>(norm_in,  (long)N);
    k_degrees<<<blocks(E), TPB, 0, stream>>>(src, dst, norm_out, norm_in, E);
    k_norms  <<<blocks(N), TPB, 0, stream>>>(norm_out, norm_in, N);

    const int gemm_blocks = (N + 127) / 128;

    // ---- Layer 0: 128 -> 128, ReLU ----
    k_prep<<<blocks((long)N * 128), TPB, 0, stream>>>(feat, norm_out, Abf, N);
    k_gemm_wmma<HID><<<gemm_blocks, TPB, 0, stream>>>(Abf, W0, T, N);
    k_zero<<<blocks((long)N * HID), TPB, 0, stream>>>(AGG, (long)N * HID);
    k_aggregate<HID><<<blocks((long)E * (HID / 4)), TPB, 0, stream>>>(T, src, dst, AGG, E);
    k_finalize<HID, true><<<blocks((long)N * HID), TPB, 0, stream>>>(AGG, norm_in, b0, H, N);

    // ---- Layer 1: 128 -> 128, ReLU ----
    k_prep<<<blocks((long)N * 128), TPB, 0, stream>>>(H, norm_out, Abf, N);
    k_gemm_wmma<HID><<<gemm_blocks, TPB, 0, stream>>>(Abf, W1, T, N);
    k_zero<<<blocks((long)N * HID), TPB, 0, stream>>>(AGG, (long)N * HID);
    k_aggregate<HID><<<blocks((long)E * (HID / 4)), TPB, 0, stream>>>(T, src, dst, AGG, E);
    k_finalize<HID, true><<<blocks((long)N * HID), TPB, 0, stream>>>(AGG, norm_in, b1, H, N);

    // ---- Layer 2: 128 -> 40, no ReLU, output to d_out ----
    k_prep<<<blocks((long)N * 128), TPB, 0, stream>>>(H, norm_out, Abf, N);
    k_gemm_wmma<CLS><<<gemm_blocks, TPB, 0, stream>>>(Abf, W2, T, N);
    k_zero<<<blocks((long)N * CLS), TPB, 0, stream>>>(AGG, (long)N * CLS);
    k_aggregate<CLS><<<blocks((long)E * (CLS / 4)), TPB, 0, stream>>>(T, src, dst, AGG, E);
    k_finalize<CLS, false><<<blocks((long)N * CLS), TPB, 0, stream>>>(AGG, norm_in, b2,
                                                                      (float*)d_out, N);
}
